// LSTMmodel_26560077758827
// MI455X (gfx1250) — matrix operand order repaired and checked by reading, not compiled, not run on hardware
//
#include <hip/hip_runtime.h>

typedef __attribute__((ext_vector_type(16))) _Float16 v16h;
typedef __attribute__((ext_vector_type(8)))  float    v8f;

#define T_SEQ 2048
#define D_IN  64
#define H     25
#define HP    32
#define NG    128   // 4 gates * 32 padded cols

#define LOG2E 1.442695040888963f

#define WMMA_F16(a, b, c) \
  __builtin_amdgcn_wmma_f32_16x16x32_f16(false, (a), false, (b), (short)0, (c), false, false)

struct Params {
  const float* x;
  const float* Wout;
  const float* bout;
  const float* Wx1[4]; const float* b1[4]; const float* Wh1[4];
  const float* Wx2[4]; const float* b2[4]; const float* Wh2[4];
  float* out;
};

// Branchless fast activations built on v_exp_f32 / v_rcp_f32 (TRANS32 pipe).
// Saturation is numerically safe: exp2->inf => rcp->0 ; exp2->0 => rcp of 1.
__device__ __forceinline__ float fast_sigmoid(float v) {
  return __builtin_amdgcn_rcpf(1.0f + __builtin_amdgcn_exp2f(v * -LOG2E));
}
__device__ __forceinline__ float fast_tanh(float v) {
  return __builtin_fmaf(2.0f,
      __builtin_amdgcn_rcpf(1.0f + __builtin_amdgcn_exp2f(v * (-2.0f * LOG2E))),
      -1.0f);
}

// A-fragment (f16 16x32, M x K) from LDS, row-major with given stride.
// Lane m=lane&15 holds row m; VGPR pair j -> K per ISA table.
__device__ __forceinline__ v16h loadA(const _Float16* base, int stride, int row, int half, int kbase) {
  v16h a;
#pragma unroll
  for (int j = 0; j < 8; ++j) {
    int K0 = kbase + (j < 4 ? 2 * j : 16 + 2 * (j - 4)) + 8 * half;
    a[2 * j]     = base[row * stride + K0];
    a[2 * j + 1] = base[row * stride + K0 + 1];
  }
  return a;
}

// B-fragment (f16 32x16, K x N) from LDS [K][NG]. Lane n=lane&15 holds col n;
// VGPR pair j -> K per ISA table (same order as loadA).
__device__ __forceinline__ v16h loadB(const _Float16* base, int col, int half, int kbase) {
  v16h b;
#pragma unroll
  for (int j = 0; j < 8; ++j) {
    int K0 = kbase + (j < 4 ? 2 * j : 16 + 2 * (j - 4)) + 8 * half;
    b[2 * j]     = base[K0 * NG + col];
    b[2 * j + 1] = base[(K0 + 1) * NG + col];
  }
  return b;
}

__global__ __launch_bounds__(256, 1) void lstm_fused(Params p) {
  __shared__ _Float16 sWx1[D_IN][NG];   // layer1 input weights, 4 gates packed
  __shared__ _Float16 sWh1[HP][NG];     // layer1 recurrent weights
  __shared__ _Float16 sWx2[HP][NG];     // layer2 input (from h1) weights
  __shared__ _Float16 sWh2[HP][NG];     // layer2 recurrent weights
  __shared__ float    sB1[NG];
  __shared__ float    sB2[NG];
  __shared__ float    sWo[HP];
  __shared__ _Float16 sX[2][16][D_IN];  // double-buffered x tile (f16)
  __shared__ _Float16 sH1[16][HP];
  __shared__ _Float16 sH2[16][HP];
  __shared__ float    sG[16][NG];       // activated gate buffer (reused per layer)
  __shared__ float    sOut[16][64];     // 64-step output staging

  const int tid  = threadIdx.x;
  const int lane = tid & 31;
  const int w    = tid >> 5;        // wave id = N-tile id (0..7)
  const int half = lane >> 4;
  const int blk  = blockIdx.x;      // 16 blocks, 16 batch rows each

  // ---------------- pack weights into LDS (f16, zero-padded 25->32) ----------------
  for (int i = tid; i < D_IN * NG; i += 256) {
    int k = i >> 7, c = i & 127, g = c >> 5, u = c & 31;
    sWx1[k][c] = (_Float16)((u < H) ? p.Wx1[g][k * H + u] : 0.0f);
  }
  for (int i = tid; i < HP * NG; i += 256) {
    int k = i >> 7, c = i & 127, g = c >> 5, u = c & 31;
    bool ok = (k < H) && (u < H);
    sWh1[k][c] = (_Float16)(ok ? p.Wh1[g][k * H + u] : 0.0f);
    sWx2[k][c] = (_Float16)(ok ? p.Wx2[g][k * H + u] : 0.0f);
    sWh2[k][c] = (_Float16)(ok ? p.Wh2[g][k * H + u] : 0.0f);
  }
  if (tid < NG) {
    int g = tid >> 5, u = tid & 31;
    sB1[tid] = (u < H) ? p.b1[g][u] : 0.0f;
    sB2[tid] = (u < H) ? p.b2[g][u] : 0.0f;
  }
  if (tid < HP) sWo[tid] = (tid < H) ? p.Wout[tid] : 0.0f;
  for (int i = tid; i < 16 * HP; i += 256) {
    sH1[i >> 5][i & 31] = (_Float16)0.0f;
    sH2[i >> 5][i & 31] = (_Float16)0.0f;
  }

  // x staging: thread -> (row, 4-float segment)
  const int xrow = tid >> 4, xseg = tid & 15;
  const float* xbase =
      p.x + ((size_t)(16 * blk + xrow) * T_SEQ) * (size_t)D_IN + 4 * xseg;
  {
    float4 v = *(const float4*)xbase;   // t = 0
    sX[0][xrow][4 * xseg + 0] = (_Float16)v.x;
    sX[0][xrow][4 * xseg + 1] = (_Float16)v.y;
    sX[0][xrow][4 * xseg + 2] = (_Float16)v.z;
    sX[0][xrow][4 * xseg + 3] = (_Float16)v.w;
  }
  __syncthreads();

  // ---------------- hoist loop-invariant B fragments + biases ----------------
  const int col = (w << 4) + (lane & 15);         // global gate column 0..127
  const v16h bx1_0 = loadB(&sWx1[0][0], col, half, 0);
  const v16h bx1_1 = loadB(&sWx1[0][0], col, half, 32);
  const v16h bh1   = loadB(&sWh1[0][0], col, half, 0);
  const v16h bx2   = loadB(&sWx2[0][0], col, half, 0);
  const v16h bh2   = loadB(&sWh2[0][0], col, half, 0);
  const float bias1 = sB1[col];
  const float bias2 = sB2[col];
  const float boutv = p.bout[0];

  // Wave-uniform, branchless gate activation:
  //   gate(v) = fma(postMul, rcp(1 + exp2(v*actScale)), postAdd)
  // sigmoid gates (i,f,o): actScale=-log2e, postMul=1, postAdd=0
  // tanh gate (c, waves 4-5): actScale=-2*log2e, postMul=2, postAdd=-1
  const bool  isTanh   = ((w >> 1) == 2);         // gate order: i,f,c(g),o
  const float actScale = isTanh ? (-2.0f * LOG2E) : (-LOG2E);
  const float postMul  = isTanh ? 2.0f : 1.0f;
  const float postAdd  = isTanh ? -1.0f : 0.0f;

  // c state in registers: thread owns elements tid and tid+256 of the 16x32 grid
  const int r0 = tid >> 5, u0 = tid & 31;
  const int r1 = 8 + (tid >> 5), u1 = tid & 31;
  float c1a = 0.0f, c1b = 0.0f, c2a = 0.0f, c2b = 0.0f;

  const int aRow = lane & 15;

  for (int t = 0; t < T_SEQ; ++t) {
    const int buf = t & 1;

    // prefetch / fetch next x tile into registers (consumed at end of iter)
    float4 xn = make_float4(0.f, 0.f, 0.f, 0.f);
    if (t + 1 < T_SEQ) xn = *(const float4*)(xbase + (size_t)(t + 1) * D_IN);
    if (t + 4 < T_SEQ) __builtin_prefetch(xbase + (size_t)(t + 4) * D_IN, 0, 1);

    // ---------------- layer 1 GEMM: gates = x@Wx1 + h1@Wh1 + b1 ----------------
    v8f acc;
#pragma unroll
    for (int r = 0; r < 8; ++r) acc[r] = bias1;
    {
      v16h ax0 = loadA(&sX[buf][0][0], D_IN, aRow, half, 0);
      v16h ax1 = loadA(&sX[buf][0][0], D_IN, aRow, half, 32);
      v16h ah  = loadA(&sH1[0][0], HP, aRow, half, 0);
      acc = WMMA_F16(ax0, bx1_0, acc);
      acc = WMMA_F16(ax1, bx1_1, acc);
      acc = WMMA_F16(ah, bh1, acc);
    }
#pragma unroll
    for (int r = 0; r < 8; ++r) {
      float e = __builtin_amdgcn_exp2f(acc[r] * actScale);
      sG[r + 8 * half][col] =
          __builtin_fmaf(postMul, __builtin_amdgcn_rcpf(1.0f + e), postAdd);
    }
    __syncthreads();

    // ---------------- layer 1 elementwise ----------------
    {
      float iv = sG[r0][u0], fv = sG[r0][32 + u0], gv = sG[r0][64 + u0], ov = sG[r0][96 + u0];
      c1a = gv * iv + fv * c1a;
      float h = ov * fast_tanh(c1a);
      sH1[r0][u0] = (_Float16)((u0 < H) ? h : 0.0f);
    }
    {
      float iv = sG[r1][u1], fv = sG[r1][32 + u1], gv = sG[r1][64 + u1], ov = sG[r1][96 + u1];
      c1b = gv * iv + fv * c1b;
      float h = ov * fast_tanh(c1b);
      sH1[r1][u1] = (_Float16)((u1 < H) ? h : 0.0f);
    }
    __syncthreads();

    // ---------------- layer 2 GEMM: gates = h1@Wx2 + h2@Wh2 + b2 ----------------
#pragma unroll
    for (int r = 0; r < 8; ++r) acc[r] = bias2;
    {
      v16h ah1 = loadA(&sH1[0][0], HP, aRow, half, 0);
      v16h ah2 = loadA(&sH2[0][0], HP, aRow, half, 0);
      acc = WMMA_F16(ah1, bx2, acc);
      acc = WMMA_F16(ah2, bh2, acc);
    }
#pragma unroll
    for (int r = 0; r < 8; ++r) {
      float e = __builtin_amdgcn_exp2f(acc[r] * actScale);
      sG[r + 8 * half][col] =
          __builtin_fmaf(postMul, __builtin_amdgcn_rcpf(1.0f + e), postAdd);
    }
    __syncthreads();

    // ---------------- layer 2 elementwise ----------------
    {
      float iv = sG[r0][u0], fv = sG[r0][32 + u0], gv = sG[r0][64 + u0], ov = sG[r0][96 + u0];
      c2a = gv * iv + fv * c2a;
      float h = ov * fast_tanh(c2a);
      sH2[r0][u0] = (_Float16)((u0 < H) ? h : 0.0f);
    }
    {
      float iv = sG[r1][u1], fv = sG[r1][32 + u1], gv = sG[r1][64 + u1], ov = sG[r1][96 + u1];
      c2b = gv * iv + fv * c2b;
      float h = ov * fast_tanh(c2b);
      sH2[r1][u1] = (_Float16)((u1 < H) ? h : 0.0f);
    }
    __syncthreads();

    // ---------------- output head: sigmoid(h2 @ Wout + bout) ----------------
    if (tid < 16) {
      float s = boutv;
#pragma unroll
      for (int k = 0; k < H; ++k) s += (float)sH2[tid][k] * sWo[k];
      sOut[tid][t & 63] = fast_sigmoid(s);
    }

    // stage next x tile into the other LDS buffer
    if (t + 1 < T_SEQ) {
      sX[buf ^ 1][xrow][4 * xseg + 0] = (_Float16)xn.x;
      sX[buf ^ 1][xrow][4 * xseg + 1] = (_Float16)xn.y;
      sX[buf ^ 1][xrow][4 * xseg + 2] = (_Float16)xn.z;
      sX[buf ^ 1][xrow][4 * xseg + 3] = (_Float16)xn.w;
    }

    // flush 64 outputs per row with coalesced b128 stores
    if ((t & 63) == 63) {
      __syncthreads();
      int tb = t - 63;
      float4 o4 = *(const float4*)&sOut[xrow][4 * xseg];
      *(float4*)(p.out + (size_t)(16 * blk + xrow) * T_SEQ + tb + 4 * xseg) = o4;
    }
    __syncthreads();
  }
}

extern "C" void kernel_launch(void* const* d_in, const int* in_sizes, int n_in,
                              void* d_out, int out_size, void* d_ws, size_t ws_size,
                              hipStream_t stream) {
  (void)in_sizes; (void)n_in; (void)d_ws; (void)ws_size; (void)out_size;
  Params p;
  p.x    = (const float*)d_in[0];
  p.Wout = (const float*)d_in[1];
  p.bout = (const float*)d_in[2];
  const int l1 = 3, l2 = 15;     // per gate: Wx, b, Wh; gate order i,f,c,o
  for (int g = 0; g < 4; ++g) {
    p.Wx1[g] = (const float*)d_in[l1 + 3 * g + 0];
    p.b1[g]  = (const float*)d_in[l1 + 3 * g + 1];
    p.Wh1[g] = (const float*)d_in[l1 + 3 * g + 2];
    p.Wx2[g] = (const float*)d_in[l2 + 3 * g + 0];
    p.b2[g]  = (const float*)d_in[l2 + 3 * g + 1];
    p.Wh2[g] = (const float*)d_in[l2 + 3 * g + 2];
  }
  p.out = (float*)d_out;
  hipLaunchKernelGGL(lstm_fused, dim3(16), dim3(256), 0, stream, p);
}
